// TRDnet_27917287424318
// MI455X (gfx1250) — matrix operand order repaired and checked by reading, not compiled, not run on hardware
//
#include <hip/hip_runtime.h>

#define SPW 8          // samples per workgroup
#define TPB 256        // 8 waves of 32

typedef __attribute__((ext_vector_type(16))) __bf16       v16bf;
typedef __attribute__((ext_vector_type(8)))  float        v8f;
typedef __attribute__((ext_vector_type(4)))  unsigned int u32x4;
typedef __attribute__((ext_vector_type(2)))  unsigned int u32x2;
typedef __attribute__((ext_vector_type(4)))  float        f32x4;

union BfFrag { u32x4 q[2]; v16bf v; };

// round-to-nearest-even f32 -> bf16 (raw u16)
__device__ __forceinline__ unsigned int f2bf(float f) {
    unsigned int u = __float_as_uint(f);
    return (u + 0x7FFFu + ((u >> 16) & 1u)) >> 16;
}

struct Smem {
    // ---- persistent weights (bf16, B-matrix row = output channel, k contiguous) ----
    unsigned short wB1[32][256];   // conv1: k = kh*16+kw
    unsigned short wB2[32][96];    // conv2: k = ic*3+kw
    unsigned short wB3[32][128];   // conv3: k = ic*4+kh*2+kw
    unsigned short wB4[32][96];    // conv4: k = ic*3+kw
    unsigned short WB5[16][32];    // fc1:   k = ic
    float scale1[32], shift1[32];  // BN folded with b1
    float b2s[32], b3s[32], b4s[32];
    float bl1s[16], W2s[16];
    float bl2s;
    int   sidx[SPW * 3];
    float h1[96][32];              // conv1 raw output, row = s*12 + ph*6 + pw
    union {
        unsigned short x[SPW][32 * 96];        // bf16 input image (conv1 A)
        struct {
            unsigned short A2[64][104];        // conv2 im2col (rows s*8+h*4+w)
            float          h2[64][32];
            unsigned short A3[32][136];        // conv3 im2col (rows s*3+w, 24 valid)
            float          h3[32][32];
            unsigned short A4[16][104];        // conv4 im2col (rows s, 8 valid)
            unsigned short h4r[16][32];        // relu(conv4+b4) bf16, rows 8..15 = 0
            float          fc1[16][16];
        } p;
    } u;
};

// vectorized f32 -> bf16 staging: n4 float4's
__device__ __forceinline__ void cvt_stage(unsigned short* dst, const float* __restrict__ src,
                                          int n4, int tid) {
    for (int i = tid; i < n4; i += TPB) {
        f32x4 v = *(const f32x4*)(src + 4 * i);
        u32x2 p;
        p[0] = f2bf(v[0]) | (f2bf(v[1]) << 16);
        p[1] = f2bf(v[2]) | (f2bf(v[3]) << 16);
        *(u32x2*)(dst + 4 * i) = p;
    }
}

// A fragment from row-major bf16 LDS row (16B-aligned rows, stride mult of 8 halfs)
// lane<16: a[0..7]=K(kk*32+0..7),  a[8..15]=K(kk*32+16..23); lane>=16: same +8
__device__ __forceinline__ v16bf load_a(const unsigned short* arow, int kk, int lane) {
    int off = (lane & 16) ? 8 : 0;
    BfFrag fa;
    const unsigned short* pa = arow + kk * 32 + off;
    fa.q[0] = *(const u32x4*)pa;
    fa.q[1] = *(const u32x4*)(pa + 16);
    return fa.v;
}

// B fragment: lane<16: b[0..7]=K(kk*32+0..7), b[8..15]=K(kk*32+16..23); lane>=16: same +8
__device__ __forceinline__ v16bf load_b(const unsigned short* brow, int kk, int lane) {
    int kb = kk * 32 + ((lane & 16) ? 8 : 0);
    BfFrag fb;
    const unsigned short* pb = brow + kb;
    fb.q[0] = *(const u32x4*)pb;
    fb.q[1] = *(const u32x4*)(pb + 16);
    return fb.v;
}

template <int KSTEPS>
__device__ __forceinline__ v8f gemm_tile(const unsigned short* arow,
                                         const unsigned short* brow, int lane) {
    v8f acc = {};
#pragma unroll
    for (int kk = 0; kk < KSTEPS; ++kk) {
        v16bf a = load_a(arow, kk, lane);
        v16bf b = load_b(brow, kk, lane);
        acc = __builtin_amdgcn_wmma_f32_16x16x32_bf16(false, a, false, b,
                                                      (short)0, acc, false, false);
    }
    return acc;
}

__global__ __launch_bounds__(TPB) void trdnet_fused(
    const int* __restrict__ idex, int nSamples,
    const float* __restrict__ E1, const float* __restrict__ E2, const float* __restrict__ E3,
    const float* __restrict__ w1, const float* __restrict__ b1,
    const float* __restrict__ w2, const float* __restrict__ b2,
    const float* __restrict__ w3, const float* __restrict__ b3,
    const float* __restrict__ w4, const float* __restrict__ b4,
    const float* __restrict__ gam, const float* __restrict__ bet,
    const float* __restrict__ mean, const float* __restrict__ var,
    const float* __restrict__ W1, const float* __restrict__ bl1,
    const float* __restrict__ W2, const float* __restrict__ bl2,
    float* __restrict__ out)
{
    __shared__ Smem sm;
    const int tid  = threadIdx.x;
    const int lane = tid & 31;
    const int wv   = tid >> 5;
    const int wg   = blockIdx.x;

    // ---------------- phase 0: stage weights (vectorized), fold BN ----------------
    cvt_stage(&sm.wB1[0][0], w1, 32 * 256 / 4, tid);
    cvt_stage(&sm.wB2[0][0], w2, 32 * 96 / 4, tid);
    cvt_stage(&sm.wB3[0][0], w3, 32 * 128 / 4, tid);
    cvt_stage(&sm.wB4[0][0], w4, 32 * 96 / 4, tid);
    cvt_stage(&sm.WB5[0][0], W1, 16 * 32 / 4, tid);
    if (tid < 32) {
        float inv = gam[tid] * rsqrtf(var[tid] + 1e-5f);
        sm.scale1[tid] = inv;
        sm.shift1[tid] = (b1[tid] - mean[tid]) * inv + bet[tid];
        sm.b2s[tid] = b2[tid]; sm.b3s[tid] = b3[tid]; sm.b4s[tid] = b4[tid];
    }
    if (tid < 16) { sm.bl1s[tid] = bl1[tid]; sm.W2s[tid] = W2[tid]; }
    if (tid == 0) sm.bl2s = bl2[0];
    if (tid < SPW * 3) {
        int gi = wg * SPW * 3 + tid;
        sm.sidx[tid] = (gi < nSamples * 3) ? idex[gi] : 0;
    }
    __syncthreads();

    // ---------------- phase 1: gather embeddings (float4) -> bf16 image ----------------
    // 256 float4 per embedding row; 3 rows per sample
    for (int i = tid; i < SPW * 768; i += TPB) {
        int s = i / 768;
        int r = i - s * 768;
        int which = r >> 8;
        int e4 = r & 255;
        int rowi = sm.sidx[s * 3 + which];
        const float* E = (which == 0) ? E1 : ((which == 1) ? E2 : E3);
        f32x4 v = *(const f32x4*)(E + (size_t)rowi * 1024 + (e4 << 2));
        int e = e4 << 2;
        int xr = e >> 5, xc = (e & 31) + which * 32;
        u32x2 p;
        p[0] = f2bf(v[0]) | (f2bf(v[1]) << 16);
        p[1] = f2bf(v[2]) | (f2bf(v[3]) << 16);
        *(u32x2*)&sm.u.x[s][xr * 96 + xc] = p;
    }
    __syncthreads();

    // ---------------- phase 2: conv1 as GEMM [96 x 32], K=256 ----------------
    for (int t = wv; t < 12; t += 8) {
        int mtile = t >> 1, n0 = (t & 1) << 4;
        int m = mtile * 16 + (lane & 15);       // row = s*12 + p
        int s = m / 12, p = m - s * 12;
        int ph = (p >= 6) ? 1 : 0, pw = p - ph * 6;
        const unsigned short* abase = &sm.u.x[s][(ph * 16) * 96 + pw * 16];
        const unsigned short* brow  = sm.wB1[n0 + (lane & 15)];
        int off = (lane & 16) ? 8 : 0;
        int kbb = (lane & 16) ? 8 : 0;
        v8f acc = {};
#pragma unroll
        for (int kk = 0; kk < 8; ++kk) {        // K-step kk covers patch rows 2kk, 2kk+1
            BfFrag fa, fb;
            const unsigned short* pa = abase + (2 * kk) * 96 + off;
            fa.q[0] = *(const u32x4*)pa;
            fa.q[1] = *(const u32x4*)(pa + 96);
            const unsigned short* pb = brow + kk * 32 + kbb;
            fb.q[0] = *(const u32x4*)pb;
            fb.q[1] = *(const u32x4*)(pb + 16);
            acc = __builtin_amdgcn_wmma_f32_16x16x32_bf16(false, fa.v, false, fb.v,
                                                          (short)0, acc, false, false);
        }
        int n = n0 + (lane & 15), mo = (lane >> 4) & 1;
#pragma unroll
        for (int vv = 0; vv < 8; ++vv)
            sm.h1[mtile * 16 + vv + 8 * mo][n] = acc[vv];
    }
    __syncthreads();

    // ---------------- phase 3: BN(+b1)+relu -> im2col for conv2 ----------------
    for (int i = tid; i < 64 * 96; i += TPB) {
        int m2 = i / 96, k = i - m2 * 96;
        int s = m2 >> 3, hp = (m2 >> 2) & 1, w = m2 & 3;
        int ic = k / 3, kw = k - ic * 3;
        int h1row = s * 12 + hp * 6 + (w + kw);
        float v = sm.h1[h1row][ic] * sm.scale1[ic] + sm.shift1[ic];
        sm.u.p.A2[m2][k] = (unsigned short)f2bf(fmaxf(v, 0.f));
    }
    __syncthreads();

    // ---------------- phase 4: conv2 GEMM [64 x 32], K=96 ----------------
    {
        int mtile = wv >> 1, n0 = (wv & 1) << 4;        // 8 tasks, one per wave
        int m = mtile * 16 + (lane & 15);
        v8f acc = gemm_tile<3>(&sm.u.p.A2[m][0], sm.wB2[n0 + (lane & 15)], lane);
        int n = n0 + (lane & 15), mo = (lane >> 4) & 1;
#pragma unroll
        for (int vv = 0; vv < 8; ++vv)
            sm.u.p.h2[mtile * 16 + vv + 8 * mo][n] = acc[vv];
    }
    __syncthreads();

    // ---------------- phase 5: relu(h2+b2) -> im2col for conv3 ----------------
    for (int i = tid; i < 24 * 128; i += TPB) {
        int m3 = i >> 7, k = i & 127;
        int s = m3 / 3, w3p = m3 - s * 3;
        int ic = k >> 2, q = k & 3, kh = q >> 1, kw = q & 1;
        float v = fmaxf(sm.u.p.h2[s * 8 + kh * 4 + (w3p + kw)][ic] + sm.b2s[ic], 0.f);
        sm.u.p.A3[m3][k] = (unsigned short)f2bf(v);
    }
    __syncthreads();

    // ---------------- phase 6: conv3 GEMM [24(+pad) x 32], K=128 ----------------
    if (wv < 4) {
        int mtile = wv >> 1, n0 = (wv & 1) << 4;
        int m = mtile * 16 + (lane & 15);
        v8f acc = gemm_tile<4>(&sm.u.p.A3[m][0], sm.wB3[n0 + (lane & 15)], lane);
        int n = n0 + (lane & 15), mo = (lane >> 4) & 1;
#pragma unroll
        for (int vv = 0; vv < 8; ++vv) {
            int mg = mtile * 16 + vv + 8 * mo;
            if (mg < 24) sm.u.p.h3[mg][n] = acc[vv];
        }
    }
    __syncthreads();

    // ---------------- phase 7: relu(h3+b3) -> im2col for conv4; zero h4r ----------------
    for (int i = tid; i < 8 * 96; i += TPB) {
        int s = i / 96, k = i - s * 96;
        int ic = k / 3, kw = k - ic * 3;
        float v = fmaxf(sm.u.p.h3[s * 3 + kw][ic] + sm.b3s[ic], 0.f);
        sm.u.p.A4[s][k] = (unsigned short)f2bf(v);
    }
    for (int i = tid; i < 16 * 32; i += TPB) ((unsigned short*)sm.u.p.h4r)[i] = 0;
    __syncthreads();

    // ---------------- phase 8: conv4 GEMM [8(+pad) x 32], K=96 ----------------
    if (wv < 2) {
        int n0 = wv << 4;
        v8f acc = gemm_tile<3>(&sm.u.p.A4[lane & 15][0], sm.wB4[n0 + (lane & 15)], lane);
        int n = n0 + (lane & 15), mo = (lane >> 4) & 1;
        if (mo == 0) {
#pragma unroll
            for (int vv = 0; vv < 8; ++vv) {
                float v = fmaxf(acc[vv] + sm.b4s[n], 0.f);
                sm.u.p.h4r[vv][n] = (unsigned short)f2bf(v);
            }
        }
    }
    __syncthreads();

    // ---------------- phase 9: FC1 as a single 16x16x32 WMMA ----------------
    if (wv == 0) {
        v8f acc = gemm_tile<1>(&sm.u.p.h4r[lane & 15][0], &sm.WB5[lane & 15][0], lane);
        int n = lane & 15, mo = (lane >> 4) & 1;
#pragma unroll
        for (int vv = 0; vv < 8; ++vv)
            sm.u.p.fc1[vv + 8 * mo][n] = fmaxf(acc[vv] + sm.bl1s[n], 0.f);
    }
    __syncthreads();

    // ---------------- phase 10: FC2 dot + store ----------------
    if (tid < SPW) {
        int sg = wg * SPW + tid;
        if (sg < nSamples) {
            float sum = sm.bl2s;
#pragma unroll
            for (int j = 0; j < 16; ++j) sum += sm.u.p.fc1[tid][j] * sm.W2s[j];
            out[sg] = sum;
        }
    }
}

extern "C" void kernel_launch(void* const* d_in, const int* in_sizes, int n_in,
                              void* d_out, int out_size, void* d_ws, size_t ws_size,
                              hipStream_t stream) {
    const int*   idex = (const int*)d_in[0];
    // d_in[1] = ranks (compile-time constant 32 here)
    const float* E1 = (const float*)d_in[2];
    const float* E2 = (const float*)d_in[3];
    const float* E3 = (const float*)d_in[4];
    const float* w1 = (const float*)d_in[5];
    const float* b1 = (const float*)d_in[6];
    const float* w2 = (const float*)d_in[7];
    const float* b2 = (const float*)d_in[8];
    const float* w3 = (const float*)d_in[9];
    const float* b3 = (const float*)d_in[10];
    const float* w4 = (const float*)d_in[11];
    const float* b4 = (const float*)d_in[12];
    const float* gam = (const float*)d_in[13];
    const float* bet = (const float*)d_in[14];
    const float* mean = (const float*)d_in[15];
    const float* var = (const float*)d_in[16];
    const float* W1 = (const float*)d_in[17];
    const float* bl1 = (const float*)d_in[18];
    const float* W2 = (const float*)d_in[19];
    const float* bl2 = (const float*)d_in[20];

    int B = in_sizes[0] / 3;
    int nwg = (B + SPW - 1) / SPW;
    trdnet_fused<<<nwg, TPB, 0, stream>>>(idex, B, E1, E2, E3,
                                          w1, b1, w2, b2, w3, b3, w4, b4,
                                          gam, bet, mean, var,
                                          W1, bl1, W2, bl2, (float*)d_out);
    (void)n_in; (void)out_size; (void)d_ws; (void)ws_size;
}
